// C2f_DCN_70076686402358
// MI455X (gfx1250) — matrix operand order repaired and checked by reading, not compiled, not run on hardware
//
#include <hip/hip_runtime.h>

typedef __attribute__((ext_vector_type(16))) _Float16 v16h;
typedef __attribute__((ext_vector_type(8)))  _Float16 v8h;
typedef __attribute__((ext_vector_type(8)))  float    v8f;

#define HWC   6400
#define WIDTH 80
#define BATCH 4
#define NB    4                              // N-subtiles per wave (16 cols each)
#define NTILES_G (BATCH * HWC / (16 * NB))   // 400 column groups; 64 cols | 6400

// ---- weight prep: f32 -> f16, rows zero-padded to Mpad = ceil16(M) ----------
__global__ void prep_w_flat_kernel(const float* __restrict__ s,
                                   _Float16* __restrict__ d, int M, int K) {
  int Mpad = (M + 15) & ~15;
  int i = blockIdx.x * blockDim.x + threadIdx.x;
  if (i >= Mpad * K) return;
  int row = i / K;
  d[i] = row < M ? (_Float16)s[i] : (_Float16)0.0f;
}

// 3x3: w[row, ci, tap] -> out[row, tap*Cin + ci] (tap-major K), rows padded
__global__ void prep_w_tapmajor_kernel(const float* __restrict__ s,
                                       _Float16* __restrict__ d, int M, int Cin) {
  int K = Cin * 9;
  int Mpad = (M + 15) & ~15;
  int i = blockIdx.x * blockDim.x + threadIdx.x;
  if (i >= Mpad * K) return;
  int row = i / K;
  int kk  = i - row * K;
  int tap = kk / Cin;
  int ci  = kk - tap * Cin;
  d[i] = row < M ? (_Float16)s[(row * Cin + ci) * 9 + tap] : (_Float16)0.0f;
}

// ---- implicit-GEMM conv + BN (+SiLU) (+residual), 16x64 tile per wave -------
// TAPS==1: K = Cin.  TAPS==9: K = 9*128 tap-major (Cin==128), 3x3 pad=1.
template<typename TB, typename TO, int TAPS, bool ACT, bool RES>
__global__ __launch_bounds__(256)
void gemm_bn_kernel(const _Float16* __restrict__ Wt,              // [Mpad,K] f16
                    const TB* __restrict__ X, int XCtot, int XcOff,
                    const float* __restrict__ bnG, const float* __restrict__ bnB,
                    const float* __restrict__ bnM, const float* __restrict__ bnV,
                    TO* __restrict__ Out, int OCtot, int OcOff,
                    const float* __restrict__ res,
                    int M, int K)
{
  const int lane = threadIdx.x & 31;
  const int wid  = threadIdx.x >> 5;
  const int mT   = (M + 15) >> 4;
  const int tile = blockIdx.x * 8 + wid;
  if (tile >= mT * NTILES_G) return;               // wave-uniform exit
  const int tm   = tile / NTILES_G;                // 8 waves/block share tm
  const int tg   = tile - tm * NTILES_G;
  const int col0 = tg * (16 * NB);
  const int bI   = col0 / HWC;
  const int pos0 = col0 - bI * HWC;
  const bool hi  = lane >= 16;
  const int  l15 = lane & 15;

  // A-operand: padded weights -> no masking, two b128 loads per K-step
  const int arow = tm * 16 + l15;
  const v8h* __restrict__ W8 =
      reinterpret_cast<const v8h*>(Wt + (size_t)arow * K);

  // B-operand: NB spatial columns per lane
  int posj[NB], hBj[NB], wBj[NB];
#pragma unroll
  for (int j = 0; j < NB; ++j) {
    posj[j] = pos0 + 16 * j + l15;
    hBj[j]  = posj[j] / WIDTH;
    wBj[j]  = posj[j] - hBj[j] * WIDTH;
  }
  const TB* __restrict__ Xb = X + (size_t)(bI * XCtot + XcOff) * HWC;

  v8f acc[NB] = {};
  for (int k0 = 0; k0 < K; k0 += 32) {
    __builtin_prefetch(reinterpret_cast<const _Float16*>(W8) + k0 + 256, 0, 1);
    // A: halves [k0+16*hi .. +7] and [k0+16*hi+8 .. +7]
    const int c0 = (k0 >> 3) + (hi ? 2 : 0);
    v8h a0 = W8[c0];
    v8h a1 = W8[c0 + 1];
    v16h av;
#pragma unroll
    for (int h = 0; h < 8; ++h) { av[h] = a0[h]; av[8 + h] = a1[h]; }

    const int kb0 = k0 + (hi ? 16 : 0);
    v16h bvv[NB];
    if (TAPS == 9) {
      const int tap = kb0 >> 7;                    // Cin == 128
      const int ciB = kb0 & 127;
      const int dyy = tap / 3 - 1;
      const int dxx = tap - (tap / 3) * 3 - 1;
      const TB* __restrict__ Xc = Xb + (size_t)ciB * HWC;
#pragma unroll
      for (int j = 0; j < NB; ++j) {
        const int hh = hBj[j] + dyy;
        const int ww = wBj[j] + dxx;
        const bool valid = (unsigned)hh < (unsigned)WIDTH &&
                           (unsigned)ww < (unsigned)WIDTH;
        const int hc = hh < 0 ? 0 : (hh > WIDTH - 1 ? WIDTH - 1 : hh);
        const int wc = ww < 0 ? 0 : (ww > WIDTH - 1 ? WIDTH - 1 : ww);
        const TB* __restrict__ p = Xc + hc * WIDTH + wc;
#pragma unroll
        for (int h = 0; h < 16; ++h) {
          _Float16 xv = (_Float16)p[(size_t)h * HWC];  // unconditional, clamped
          bvv[j][h] = valid ? xv : (_Float16)0.0f;     // cndmask, no branches
        }
      }
    } else {
      const TB* __restrict__ p0 = Xb + (size_t)kb0 * HWC;
#pragma unroll
      for (int j = 0; j < NB; ++j)
#pragma unroll
        for (int h = 0; h < 16; ++h)
          bvv[j][h] = (_Float16)p0[(size_t)h * HWC + posj[j]];
    }
#pragma unroll
    for (int j = 0; j < NB; ++j)                   // 4 independent WMMA chains
      acc[j] = __builtin_amdgcn_wmma_f32_16x16x32_f16(false, av, false, bvv[j],
                                                      (short)0, acc[j], false, false);
  }

  // Epilogue: C/D layout -> VGPR r holds row r (+8 for hi lanes), col = l15
#pragma unroll
  for (int r = 0; r < 8; ++r) {
    int row = tm * 16 + r + (hi ? 8 : 0);
    if (row < M) {
      float sc = bnG[row] * rsqrtf(bnV[row] + 1e-5f);
      float sh = bnB[row] - bnM[row] * sc;
#pragma unroll
      for (int j = 0; j < NB; ++j) {
        float y = acc[j][r] * sc + sh;
        if (ACT) y = y / (1.0f + __expf(-y));      // SiLU
        if (RES) y += res[(size_t)(bI * M + row) * HWC + posj[j]];
        Out[(size_t)(bI * OCtot + OcOff + row) * HWC + posj[j]] = (TO)y;
      }
    }
  }
}

// ---- deformable bilinear gather --------------------------------------------
__global__ __launch_bounds__(256)
void dcn_gather_kernel(const _Float16* __restrict__ X,   // [B,128,HW] f16
                       const float* __restrict__ OFF,    // [B,18,HW]
                       _Float16* __restrict__ VAL)       // [B,1152,HW], K=ci*9+tap
{
  int idx = blockIdx.x * blockDim.x + threadIdx.x;
  if (idx >= BATCH * 9 * HWC) return;
  int pos = idx % HWC;
  int t   = idx / HWC;
  int k   = t % 9;
  int bI  = t / 9;
  int h = pos / WIDTH;
  int w = pos - h * WIDTH;
  float dy = OFF[(size_t)(bI * 18 + 2 * k    ) * HWC + pos];
  float dx = OFF[(size_t)(bI * 18 + 2 * k + 1) * HWC + pos];
  float py = dy + (float)(h + k / 3 - 1);
  float px = dx + (float)(w + k % 3 - 1);
  float fy = floorf(py), fx = floorf(px);
  float wy1 = py - fy, wy0 = 1.0f - wy1;
  float wx1 = px - fx, wx0 = 1.0f - wx1;
  int y0 = (int)fy, x0 = (int)fx;
  int yy[4] = {y0, y0, y0 + 1, y0 + 1};
  int xx[4] = {x0, x0 + 1, x0, x0 + 1};
  float wb[4] = {wy0 * wx0, wy0 * wx1, wy1 * wx0, wy1 * wx1};
  int   o[4]; float wq[4];
#pragma unroll
  for (int c = 0; c < 4; ++c) {
    bool v = yy[c] >= 0 && yy[c] <= WIDTH - 1 && xx[c] >= 0 && xx[c] <= WIDTH - 1;
    int yi = yy[c] < 0 ? 0 : (yy[c] > WIDTH - 1 ? WIDTH - 1 : yy[c]);
    int xi = xx[c] < 0 ? 0 : (xx[c] > WIDTH - 1 ? WIDTH - 1 : xx[c]);
    o[c]  = yi * WIDTH + xi;
    wq[c] = v ? wb[c] : 0.0f;
  }
  const _Float16* Xb = X + (size_t)bI * 128 * HWC;
  _Float16* Vb = VAL + ((size_t)bI * 1152 + k) * HWC + pos;
  for (int ci = 0; ci < 128; ++ci) {
    const _Float16* xc = Xb + (size_t)ci * HWC;
    float v = wq[0] * (float)xc[o[0]] + wq[1] * (float)xc[o[1]]
            + wq[2] * (float)xc[o[2]] + wq[3] * (float)xc[o[3]];
    Vb[(size_t)ci * 9 * HWC] = (_Float16)v;
  }
}

extern "C" void kernel_launch(void* const* d_in, const int* in_sizes, int n_in,
                              void* d_out, int out_size, void* d_ws, size_t ws_size,
                              hipStream_t stream)
{
  (void)in_sizes; (void)n_in; (void)out_size; (void)ws_size;
  // Input order (setup_inputs insertion order, params flattened recursively):
  // 0:x 1:cv1_w 2-5:cv1_bn | per block (base=6,21): +0 conv_w +1..4 conv_bn
  // +5 off_w +6..9 off_bn +10 dcn_w +11..14 bn2 | 36:cv2_w 37-40:cv2_bn
  char* ws = (char*)d_ws;
  size_t off = 0;
  auto alloc = [&](size_t bytes) -> void* {
    void* p = ws + off;
    off = (off + bytes + 255) & ~(size_t)255;
    return p;
  };
  _Float16* cat  = (_Float16*)alloc((size_t)BATCH * 512  * HWC * 2); // y0|y1|y2
  _Float16* x1   = (_Float16*)alloc((size_t)BATCH * 128  * HWC * 2);
  float*    offb = (float*)   alloc((size_t)BATCH * 18   * HWC * 4);
  _Float16* val  = (_Float16*)alloc((size_t)BATCH * 1152 * HWC * 2);

  _Float16* wCv1  = (_Float16*)alloc((size_t)256 * 256  * 2);
  _Float16* wConv[2]; _Float16* wOff[2]; _Float16* wDcn[2];
  for (int b = 0; b < 2; ++b) {
    wConv[b] = (_Float16*)alloc((size_t)128 * 1152 * 2);
    wOff[b]  = (_Float16*)alloc((size_t)32  * 1152 * 2);
    wDcn[b]  = (_Float16*)alloc((size_t)128 * 1152 * 2);
  }
  _Float16* wCv2 = (_Float16*)alloc((size_t)256 * 512 * 2);

  auto prepFlat = [&](const void* s, _Float16* d, int M, int K) {
    int n = ((M + 15) & ~15) * K;
    prep_w_flat_kernel<<<(n + 255) / 256, 256, 0, stream>>>((const float*)s, d, M, K);
  };
  auto prepTap = [&](const void* s, _Float16* d, int M, int Cin) {
    int n = ((M + 15) & ~15) * Cin * 9;
    prep_w_tapmajor_kernel<<<(n + 255) / 256, 256, 0, stream>>>((const float*)s, d, M, Cin);
  };
  prepFlat(d_in[1],  wCv1, 256, 256);
  prepTap (d_in[6],  wConv[0], 128, 128);
  prepTap (d_in[11], wOff[0],  18,  128);
  prepFlat(d_in[16], wDcn[0],  128, 1152);
  prepTap (d_in[21], wConv[1], 128, 128);
  prepTap (d_in[26], wOff[1],  18,  128);
  prepFlat(d_in[31], wDcn[1],  128, 1152);
  prepFlat(d_in[36], wCv2, 256, 512);

  auto F = [&](int i) { return (const float*)d_in[i]; };
  const float* x = (const float*)d_in[0];
  float* out = (float*)d_out;
  auto grid = [](int M) {
    int mT = (M + 15) / 16;
    return dim3((unsigned)((mT * NTILES_G + 7) / 8));
  };

  // cv1: 1x1 256->256, BN+SiLU, from f32 x -> cat ch[0..255]
  gemm_bn_kernel<float, _Float16, 1, true, false><<<grid(256), 256, 0, stream>>>(
      wCv1, x, 256, 0, F(2), F(3), F(4), F(5), cat, 512, 0, nullptr, 256, 256);

  for (int blk = 0; blk < 2; ++blk) {
    int base = 6 + blk * 15;
    // 3x3 conv 128->128, BN only; input = cat ch[128+128*blk ..] (ys[-1])
    gemm_bn_kernel<_Float16, _Float16, 9, false, false><<<grid(128), 256, 0, stream>>>(
        wConv[blk], cat, 512, 128 + 128 * blk,
        F(base + 1), F(base + 2), F(base + 3), F(base + 4),
        x1, 128, 0, nullptr, 128, 1152);
    // 3x3 offset conv 128->18, BN+SiLU -> f32 offsets
    gemm_bn_kernel<_Float16, float, 9, true, false><<<grid(18), 256, 0, stream>>>(
        wOff[blk], x1, 128, 0,
        F(base + 6), F(base + 7), F(base + 8), F(base + 9),
        offb, 18, 0, nullptr, 18, 1152);
    // deformable bilinear gather -> val[B,1152,HW]
    dcn_gather_kernel<<<(BATCH * 9 * HWC + 255) / 256, 256, 0, stream>>>(x1, offb, val);
    // DCN as K=1152 GEMM (ci-major, matches val), bn2+SiLU -> cat ch[256+128*blk ..]
    gemm_bn_kernel<_Float16, _Float16, 1, true, false><<<grid(128), 256, 0, stream>>>(
        wDcn[blk], val, 1152, 0,
        F(base + 11), F(base + 12), F(base + 13), F(base + 14),
        cat, 512, 256 + 128 * blk, nullptr, 128, 1152);
  }

  // cv2: 1x1 512->256, BN+SiLU, + residual x -> f32 out
  gemm_bn_kernel<_Float16, float, 1, true, true><<<grid(256), 256, 0, stream>>>(
      wCv2, cat, 512, 0, F(37), F(38), F(39), F(40), out, 256, 0, x, 256, 512);
}
